// Attention_40973988004369
// MI455X (gfx1250) — matrix operand order repaired and checked by reading, not compiled, not run on hardware
//
#include <hip/hip_runtime.h>
#include <hip/hip_bf16.h>

typedef _Float16 v16h __attribute__((ext_vector_type(16)));
typedef _Float16 v8h  __attribute__((ext_vector_type(8)));
typedef float    v8f  __attribute__((ext_vector_type(8)));

#define WMMA16(a, b, c) __builtin_amdgcn_wmma_f32_16x16x32_f16( \
    false, (a), false, (b), (short)0, (c), false, false)

#define BATCH   16
#define SEQ     1024
#define HEADS   8
#define HD      32
#define DIMC    256
#define SCALE_Q 0.17677669529663687f  // 32^-0.5

// Load an A-matrix (16x32 f16) per-lane fragment: row base pointer must already
// include (+ row*stride + 16*(lane>>4)); chunks at +0 and +8.
__device__ __forceinline__ v16h ld_a16(const _Float16* __restrict__ p) {
    v8h lo = *(const v8h*)(p);
    v8h hi = *(const v8h*)(p + 8);
    v16h r;
#pragma unroll
    for (int i = 0; i < 8; ++i) { r[i] = lo[i]; r[i + 8] = hi[i]; }
    return r;
}

// ---------------- conversion kernels ----------------
__global__ void cvt_f32_f16(const float* __restrict__ s, _Float16* __restrict__ d, int n) {
    int i = blockIdx.x * blockDim.x + threadIdx.x;
    if (i < n) d[i] = (_Float16)s[i];
}

// biasx[h][n][m] = bias_table[rel[n*1024+m]*8 + h]  (f16, 8M elements)
__global__ void bias_gather(const float* __restrict__ bt, const int* __restrict__ rel,
                            _Float16* __restrict__ bx) {
    int i = blockIdx.x * blockDim.x + threadIdx.x;
    int m = i & (SEQ - 1);
    int n = (i >> 10) & (SEQ - 1);
    int h = i >> 20;
    bx[i] = (_Float16)bt[rel[n * SEQ + m] * HEADS + h];
}

// ---------------- QKV projection ----------------
// Y = x @ qkv_w^T  (M=16384, N=768, K=256), scatter into q16 (scaled), k16, vt16.
__global__ __launch_bounds__(128)
void qkv_gemm(const _Float16* __restrict__ x16, const _Float16* __restrict__ w16,
              _Float16* __restrict__ q16, _Float16* __restrict__ k16,
              _Float16* __restrict__ vt16) {
    int lane = threadIdx.x & 31, wave = threadIdx.x >> 5;
    int lo = lane & 15, hi = lane >> 4;
    int tn = blockIdx.x;                 // 0..47
    int tm = blockIdx.y * 4 + wave;      // 0..1023
    const _Float16* arow = x16 + (size_t)(tm * 16 + lo) * DIMC + 16 * hi;
    const _Float16* brow = w16 + (size_t)(tn * 16 + lo) * DIMC + 16 * hi;
    v8f c = {};
#pragma unroll
    for (int ks = 0; ks < 8; ++ks) {
        v16h a = ld_a16(arow + ks * 32);
        v16h b = *(const v16h*)(brow + ks * 32);
        c = WMMA16(a, b, c);
    }
    int j0 = tn * 16;
    int s = j0 >> 8, h = (j0 >> 5) & 7, d = (j0 & 31) + lo;
#pragma unroll
    for (int r = 0; r < 8; ++r) {
        int mrow = tm * 16 + r + 8 * hi;      // global token (over B*N)
        int bb = mrow >> 10, t = mrow & (SEQ - 1);
        float v = c[r];
        size_t bh = (size_t)bb * HEADS + h;
        if (s == 0)      q16[(bh * SEQ + t) * HD + d] = (_Float16)(v * SCALE_Q);
        else if (s == 1) k16[(bh * SEQ + t) * HD + d] = (_Float16)v;
        else             vt16[(bh * HD + d) * SEQ + t] = (_Float16)v;
    }
}

// ---------------- fused flash attention ----------------
// One wave = 16 query rows of one (b,h); loop 32-key blocks with online softmax.
__global__ __launch_bounds__(256)
void attn_kernel(const _Float16* __restrict__ q16, const _Float16* __restrict__ k16,
                 const _Float16* __restrict__ vt16, const _Float16* __restrict__ bx,
                 _Float16* __restrict__ xo16) {
    __shared__ __align__(16) _Float16 pbuf[8 * 16 * 40];   // 8 waves * 16x32 (stride 40)
    int lane = threadIdx.x & 31, wave = threadIdx.x >> 5;
    int lo = lane & 15, hi = lane >> 4;
    int bh = blockIdx.x >> 3, qc = blockIdx.x & 7;
    int b = bh >> 3, h = bh & 7;
    int qbase = qc * 128 + wave * 16;
    _Float16* pb = pbuf + wave * 16 * 40;

    const _Float16* krows = k16 + (size_t)bh * SEQ * HD;
    const _Float16* vrows = vt16 + (size_t)bh * HD * SEQ;
    v16h qa = ld_a16(q16 + ((size_t)bh * SEQ + qbase + lo) * HD + 16 * hi);

    v8f o0 = {}, o1 = {};
    float mrow[8], lrow[8];
#pragma unroll
    for (int r = 0; r < 8; ++r) { mrow[r] = -3.0e38f; lrow[r] = 0.0f; }
    v8f zero = {};

    for (int j = 0; j < SEQ; j += 32) {
        // S = Q K^T for two 16-key tiles (K-dim = head_dim = 32)
        v16h kb0 = *(const v16h*)(krows + (size_t)(j + lo) * HD + 16 * hi);
        v16h kb1 = *(const v16h*)(krows + (size_t)(j + 16 + lo) * HD + 16 * hi);
        v8f s0 = WMMA16(qa, kb0, zero);
        v8f s1 = WMMA16(qa, kb1, zero);

        // add relative-position bias (precomputed, coalesced over lanes)
#pragma unroll
        for (int r = 0; r < 8; ++r) {
            int qrow = qbase + r + 8 * hi;
            const _Float16* bp = bx + ((size_t)h * SEQ + qrow) * SEQ + j + lo;
            s0[r] += (float)bp[0];
            s1[r] += (float)bp[16];
        }

        // online softmax per row (rows live in 16-lane groups of C layout)
#pragma unroll
        for (int r = 0; r < 8; ++r) {
            float v = fmaxf(s0[r], s1[r]);
            v = fmaxf(v, __shfl_xor(v, 1));
            v = fmaxf(v, __shfl_xor(v, 2));
            v = fmaxf(v, __shfl_xor(v, 4));
            v = fmaxf(v, __shfl_xor(v, 8));
            float nm = fmaxf(mrow[r], v);
            float sc = __expf(mrow[r] - nm);
            mrow[r] = nm;
            float p0 = __expf(s0[r] - nm);
            float p1 = __expf(s1[r] - nm);
            float ts = p0 + p1;
            ts += __shfl_xor(ts, 1);
            ts += __shfl_xor(ts, 2);
            ts += __shfl_xor(ts, 4);
            ts += __shfl_xor(ts, 8);
            lrow[r] = lrow[r] * sc + ts;
            o0[r] *= sc;
            o1[r] *= sc;
            int m = r + 8 * hi;                 // C-layout row
            pb[m * 40 + lo]      = (_Float16)p0;  // transpose staging via LDS
            pb[m * 40 + 16 + lo] = (_Float16)p1;
        }
        __builtin_amdgcn_wave_barrier();
        asm volatile("s_wait_dscnt 0" ::: "memory");

        // re-read P in A-matrix layout (same wave; DS ops are in-order)
        v16h pa;
        {
            const _Float16* pp = pb + lo * 40 + 16 * hi;
            v8h plo = *(const v8h*)(pp);
            v8h phi = *(const v8h*)(pp + 8);
#pragma unroll
            for (int i = 0; i < 8; ++i) { pa[i] = plo[i]; pa[i + 8] = phi[i]; }
        }
        __builtin_amdgcn_wave_barrier();

        // O += P @ V  (K-dim = 32 keys; B from V^T rows, contiguous)
        v16h vb0 = *(const v16h*)(vrows + (size_t)lo * SEQ + j + 16 * hi);
        v16h vb1 = *(const v16h*)(vrows + (size_t)(16 + lo) * SEQ + j + 16 * hi);
        o0 = WMMA16(pa, vb0, o0);
        o1 = WMMA16(pa, vb1, o1);
    }

    // normalize and store [b][token][h*32 + d] as f16 for the proj GEMM
#pragma unroll
    for (int r = 0; r < 8; ++r) {
        float inv = 1.0f / lrow[r];
        int token = qbase + r + 8 * hi;
        size_t base = ((size_t)b * SEQ + token) * DIMC + h * HD;
        xo16[base + lo]      = (_Float16)(o0[r] * inv);
        xo16[base + 16 + lo] = (_Float16)(o1[r] * inv);
    }
}

// ---------------- output projection ----------------
__global__ __launch_bounds__(128)
void proj_gemm(const _Float16* __restrict__ xo16, const _Float16* __restrict__ w16,
               const float* __restrict__ pbias, float* __restrict__ out) {
    int lane = threadIdx.x & 31, wave = threadIdx.x >> 5;
    int lo = lane & 15, hi = lane >> 4;
    int tn = blockIdx.x;                 // 0..15
    int tm = blockIdx.y * 4 + wave;      // 0..1023
    const _Float16* arow = xo16 + (size_t)(tm * 16 + lo) * DIMC + 16 * hi;
    const _Float16* brow = w16 + (size_t)(tn * 16 + lo) * DIMC + 16 * hi;
    v8f c = {};
#pragma unroll
    for (int ks = 0; ks < 8; ++ks) {
        v16h a = ld_a16(arow + ks * 32);
        v16h b = *(const v16h*)(brow + ks * 32);
        c = WMMA16(a, b, c);
    }
    float bias = pbias[tn * 16 + lo];
#pragma unroll
    for (int r = 0; r < 8; ++r) {
        int mrow = tm * 16 + r + 8 * hi;
        out[(size_t)mrow * DIMC + tn * 16 + lo] = c[r] + bias;
    }
}

extern "C" void kernel_launch(void* const* d_in, const int* in_sizes, int n_in,
                              void* d_out, int out_size, void* d_ws, size_t ws_size,
                              hipStream_t stream) {
    const float* x      = (const float*)d_in[0];   // [16,1024,256]
    const float* qkv_w  = (const float*)d_in[1];   // [768,256]
    const float* proj_w = (const float*)d_in[2];   // [256,256]
    const float* proj_b = (const float*)d_in[3];   // [256]
    const float* bt     = (const float*)d_in[4];   // [3969,8]
    const int*   rel    = (const int*)d_in[5];     // [1024,1024]
    float* out = (float*)d_out;

    char* ws = (char*)d_ws;
    _Float16* q16  = (_Float16*)(ws);                               // 8 MiB
    _Float16* k16  = (_Float16*)(ws + (8u  << 20));                 // 8 MiB
    _Float16* vt16 = (_Float16*)(ws + (16u << 20));                 // 8 MiB
    _Float16* xo16 = (_Float16*)(ws + (24u << 20));                 // 8 MiB
    _Float16* x16  = (_Float16*)(ws + (32u << 20));                 // 8 MiB
    _Float16* w1   = (_Float16*)(ws + (40u << 20));                 // 384 KiB
    _Float16* w2   = (_Float16*)(ws + (40u << 20) + (1u << 19));    // 128 KiB
    _Float16* bx   = (_Float16*)(ws + (41u << 20));                 // 16 MiB

    cvt_f32_f16<<<16384, 256, 0, stream>>>(x, x16, BATCH * SEQ * DIMC);
    cvt_f32_f16<<<768,   256, 0, stream>>>(qkv_w, w1, 3 * DIMC * DIMC);
    cvt_f32_f16<<<256,   256, 0, stream>>>(proj_w, w2, DIMC * DIMC);
    bias_gather<<<32768, 256, 0, stream>>>(bt, rel, bx);

    qkv_gemm<<<dim3(48, 256), 128, 0, stream>>>(x16, w1, q16, k16, vt16);
    attn_kernel<<<BATCH * HEADS * 8, 256, 0, stream>>>(q16, k16, vt16, bx, xo16);
    proj_gemm<<<dim3(16, 256), 128, 0, stream>>>(xo16, w2, proj_b, out);
}
